// CCGGenerator_19963007991868
// MI455X (gfx1250) — matrix operand order repaired and checked by reading, not compiled, not run on hardware
//
#include <hip/hip_runtime.h>
#include <hip/hip_bf16.h>
#include <stdint.h>

// Problem constants (from reference)
#define BQ    128      // batch
#define NSEQ  1024     // encoder sequence length
#define HID   512      // hidden
#define CIN   32       // condition channels
#define TOUT  24       // decoder steps
#define G4H   2048     // 4*H gate width

#define NBLOCKS   64
#define NTHREADS  128   // 4 wave32 waves per block
#define SMEM_BYTES (64*1024 + 64)   // 64KB weight cache + reduction scratch

typedef __attribute__((ext_vector_type(16))) __bf16 v16bf;
typedef __attribute__((ext_vector_type(8)))  float  v8f;

union FragB16 { v16bf bf; unsigned int u[8]; };

struct Ptrs {
  const float* cond; const float* Wih_enc; const float* Whh_enc;
  const float* bih_enc; const float* bhh_enc;
  const float* Wih_cell; const float* bih_cell; const float* bhh_cell;
  const float* W_out; const float* b_out;
  unsigned int* bar_cnt; unsigned int* bar_gen;
  float* bias_enc; float* bias_cell;
  __bf16* hbuf;                 // [2][BQ][HID] double buffer (single pointer!)
  __bf16* cond_bf; __bf16* whh_bf; __bf16* wih_enc_bf; __bf16* wih_cell_bf;
  float* scores; float* fin; __bf16* cl;
  float* out;
};

// ---------------- device helpers ----------------

__device__ __forceinline__ float sigf(float x) { return 1.f / (1.f + __expf(-x)); }
__device__ __forceinline__ float tanhfast(float x) { return 1.f - 2.f / (__expf(2.f * x) + 1.f); }

__device__ __forceinline__ float2 bf2f2(unsigned int u) {
  float2 r;
  r.x = __uint_as_float(u << 16);
  r.y = __uint_as_float(u & 0xffff0000u);
  return r;
}

__device__ __forceinline__ FragB16 pack_frag(uint4 a, uint4 b) {
  FragB16 f;
  f.u[0] = a.x; f.u[1] = a.y; f.u[2] = a.z; f.u[3] = a.w;
  f.u[4] = b.x; f.u[5] = b.y; f.u[6] = b.z; f.u[7] = b.w;
  return f;
}

// A-operand (16x32 bf16, row-major row of this lane's M), contiguous layout:
// dwords 8*hi .. 8*hi+7  ->  two aligned uint4 loads.
__device__ __forceinline__ FragB16 load_fragA_row(const __bf16* rowptr, int hi) {
  const uint4* p4 = (const uint4*)rowptr;
  return pack_frag(p4[2 * hi], p4[2 * hi + 1]);
}

// B-operand (32x16 bf16, W row-major row n = lane&15): dwords 8*hi .. 8*hi+7.
__device__ __forceinline__ FragB16 load_fragB_row(const __bf16* rowptr, int hi) {
  const uint4* p4 = (const uint4*)rowptr;
  return pack_frag(p4[2 * hi], p4[2 * hi + 1]);
}

// LDS fragment cache read: per-lane base + compile-time fragment offset
// -> folds into ds_load_b128 immediate offsets, zero per-iteration address math.
__device__ __forceinline__ FragB16 load_fragB_lds(const uint4* lane_base, int fragIdx) {
  return pack_frag(lane_base[fragIdx * 64], lane_base[fragIdx * 64 + 1]);
}

__device__ __forceinline__ v8f wmma_bf16(FragB16 a, FragB16 b, v8f c) {
  return __builtin_amdgcn_wmma_f32_16x16x32_bf16(false, a.bf, false, b.bf,
                                                 (short)0, c, false, false);
}

__device__ __forceinline__ void grid_barrier(unsigned int* cnt, unsigned int* gen) {
  __threadfence();
  __syncthreads();
  if (threadIdx.x == 0) {
    volatile unsigned int* vgen = gen;
    unsigned int g = *vgen;
    if (atomicAdd(cnt, 1u) == (unsigned)(NBLOCKS - 1)) {
      atomicExch(cnt, 0u);
      __threadfence();
      atomicAdd(gen, 1u);
    } else {
      while (*vgen == g) __builtin_amdgcn_s_sleep(1);
    }
  }
  __syncthreads();
}

__device__ __forceinline__ float block_reduce_max(float v, float* red) {
#pragma unroll
  for (int o = 16; o > 0; o >>= 1) v = fmaxf(v, __shfl_xor(v, o, 32));
  int w = threadIdx.x >> 5;
  if ((threadIdx.x & 31) == 0) red[w] = v;
  __syncthreads();
  v = fmaxf(fmaxf(red[0], red[1]), fmaxf(red[2], red[3]));
  __syncthreads();
  return v;
}

__device__ __forceinline__ float block_reduce_sum(float v, float* red) {
#pragma unroll
  for (int o = 16; o > 0; o >>= 1) v += __shfl_xor(v, o, 32);
  int w = threadIdx.x >> 5;
  if ((threadIdx.x & 31) == 0) red[w] = v;
  __syncthreads();
  v = red[0] + red[1] + red[2] + red[3];
  __syncthreads();
  return v;
}

__device__ __forceinline__ float dot8(uint4 a, uint4 b, float s) {
  float2 a0 = bf2f2(a.x), b0 = bf2f2(b.x);
  s = fmaf(a0.x, b0.x, s); s = fmaf(a0.y, b0.y, s);
  float2 a1 = bf2f2(a.y), b1 = bf2f2(b.y);
  s = fmaf(a1.x, b1.x, s); s = fmaf(a1.y, b1.y, s);
  float2 a2 = bf2f2(a.z), b2 = bf2f2(b.z);
  s = fmaf(a2.x, b2.x, s); s = fmaf(a2.y, b2.y, s);
  float2 a3 = bf2f2(a.w), b3 = bf2f2(b.w);
  s = fmaf(a3.x, b3.x, s); s = fmaf(a3.y, b3.y, s);
  return s;
}

// Cache recurrent weight columns for this block's jt into LDS, fragment-major:
// per (gate gi, k-tile kt), lane slot = 32 contiguous bytes (two b128 ops).
__device__ void fill_lds_w(unsigned int* lds_w, const __bf16* Wsrc, int jt, int tid) {
  for (int idx = tid; idx < 4 * 16 * 32; idx += NTHREADS) {
    int f = idx >> 5, ln = idx & 31;
    int gi = f >> 4, kt = f & 15;
    int l15 = ln & 15, h2 = ln >> 4;
    const uint4* src =
        (const uint4*)(Wsrc + ((size_t)(gi * HID + jt * 16 + l15)) * HID + kt * 32);
    uint4* dst = (uint4*)(lds_w + (size_t)idx * 8);
    dst[0] = src[2 * h2];
    dst[1] = src[2 * h2 + 1];
  }
}

// ---------------- kernels ----------------

__global__ void ccg_init_kernel(Ptrs p) {
  size_t gt = (size_t)blockIdx.x * blockDim.x + threadIdx.x;
  size_t stride = (size_t)gridDim.x * blockDim.x;
  for (size_t i = gt; i < (size_t)G4H * HID; i += stride) p.whh_bf[i] = (__bf16)p.Whh_enc[i];
  for (size_t i = gt; i < (size_t)G4H * CIN; i += stride) p.wih_enc_bf[i] = (__bf16)p.Wih_enc[i];
  for (size_t i = gt; i < (size_t)G4H * HID; i += stride) p.wih_cell_bf[i] = (__bf16)p.Wih_cell[i];
  for (size_t i = gt; i < (size_t)BQ * NSEQ * CIN; i += stride) p.cond_bf[i] = (__bf16)p.cond[i];
  for (size_t i = gt; i < (size_t)G4H; i += stride) {
    p.bias_enc[i] = p.bih_enc[i] + p.bhh_enc[i];
    p.bias_cell[i] = p.bih_cell[i] + p.bhh_cell[i];
  }
  for (size_t i = gt; i < (size_t)BQ * HID; i += stride) p.hbuf[i] = (__bf16)0.f;
  if (gt == 0) { *p.bar_cnt = 0u; *p.bar_gen = 0u; }
}

__global__ void __launch_bounds__(NTHREADS)
ccg_persistent_kernel(Ptrs p) {
  extern __shared__ unsigned int smem_u32[];
  unsigned int* lds_w = smem_u32;                       // 16384 dwords = 64KB
  float* lds_red = (float*)(smem_u32 + 16384);          // reduction scratch

  const int tid = threadIdx.x;
  const int wave = tid >> 5, lane = tid & 31;
  const int lane15 = lane & 15, hi = lane >> 4;
  const int jt = blockIdx.x & 31;                       // hidden tile 0..31
  const int bt = ((blockIdx.x >> 5) << 2) + wave;       // batch tile 0..7
  const int jcol = jt * 16 + lane15;                    // this lane's N column
  const int brow0 = bt * 16;

  const uint4* lds_lane = (const uint4*)lds_w + lane * 2;   // per-lane fragment base

  // Wih_enc B fragments (K=32 == one WMMA) live in registers for the whole encoder
  FragB16 wihB[4];
#pragma unroll
  for (int gi = 0; gi < 4; ++gi)
    wihB[gi] = load_fragB_row(p.wih_enc_bf + (size_t)(gi * HID + jcol) * CIN, hi);

  fill_lds_w(lds_w, p.whh_bf, jt, tid);
  __syncthreads();

  // loop-invariant hoists
  float bias_v[4];
#pragma unroll
  for (int gi = 0; gi < 4; ++gi) bias_v[gi] = p.bias_enc[gi * HID + jcol];

  const __bf16* cond_row = p.cond_bf + (size_t)(brow0 + lane15) * NSEQ * CIN;
  const size_t arow_off = (size_t)(brow0 + lane15) * HID;   // A-row offset in h buffers

  // cell state lives entirely in VGPRs: this wave exclusively owns its (b,j) tile
  float creg[8];
#pragma unroll
  for (int r = 0; r < 8; ++r) creg[r] = 0.f;

  int cur = 0;

  // ---------------- encoder LSTM over N=1024 steps ----------------
  for (int n = 0; n < NSEQ; ++n) {
    v8f acc[4];
#pragma unroll
    for (int gi = 0; gi < 4; ++gi) {
#pragma unroll
      for (int r = 0; r < 8; ++r) acc[gi][r] = bias_v[gi];
    }

    // input contribution: cond[b, n, :] @ Wih^T   (K = 32)
    {
      FragB16 a = load_fragA_row(cond_row + (size_t)n * CIN, hi);
      __builtin_prefetch(cond_row + (size_t)(n + 1) * CIN, 0, 3);  // global_prefetch_b8
#pragma unroll
      for (int gi = 0; gi < 4; ++gi) acc[gi] = wmma_bf16(a, wihB[gi], acc[gi]);
    }

    // recurrent contribution: h @ Whh^T   (K = 512, from LDS weight cache)
    const __bf16* hrow = p.hbuf + (size_t)cur * (BQ * HID) + arow_off;
    for (int kt = 0; kt < 16; ++kt) {
      FragB16 a = load_fragA_row(hrow + kt * 32, hi);
#pragma unroll
      for (int gi = 0; gi < 4; ++gi) {
        FragB16 b = load_fragB_lds(lds_lane, gi * 16 + kt);
        acc[gi] = wmma_bf16(a, b, acc[gi]);
      }
    }

    // pointwise LSTM update on the accumulator fragments
    __bf16* hout = p.hbuf + (size_t)(cur ^ 1) * (BQ * HID);
#pragma unroll
    for (int r = 0; r < 8; ++r) {
      int b = brow0 + r + (hi << 3);
      float iv = sigf(acc[0][r]);
      float fv = sigf(acc[1][r]);
      float gv = tanhfast(acc[2][r]);
      float ov = sigf(acc[3][r]);
      float cn = fmaf(fv, creg[r], iv * gv);
      creg[r] = cn;
      float hv = ov * tanhfast(cn);
      __bf16 hx = (__bf16)hv;
      hout[(size_t)b * HID + jcol] = hx;
      p.cl[((size_t)b * NSEQ + n) * HID + jcol] = hx;   // bf16 KV cache (L2-resident)
    }

    grid_barrier(p.bar_cnt, p.bar_gen);
    cur ^= 1;
  }

  // ---------------- decoder: swap LDS weights to Wih_cell ----------------
  __syncthreads();
  fill_lds_w(lds_w, p.wih_cell_bf, jt, tid);
  __syncthreads();

#pragma unroll
  for (int gi = 0; gi < 4; ++gi) bias_v[gi] = p.bias_cell[gi * HID + jcol];

  const int gtid = blockIdx.x * NTHREADS + tid;

  for (int t = 0; t < TOUT; ++t) {
    // gates = h_prev @ Wih_cell^T + bias (h0=c0=0 inside the cell)
    v8f acc[4];
#pragma unroll
    for (int gi = 0; gi < 4; ++gi) {
#pragma unroll
      for (int r = 0; r < 8; ++r) acc[gi][r] = bias_v[gi];
    }
    const __bf16* hrow = p.hbuf + (size_t)cur * (BQ * HID) + arow_off;
    for (int kt = 0; kt < 16; ++kt) {
      FragB16 a = load_fragA_row(hrow + kt * 32, hi);
#pragma unroll
      for (int gi = 0; gi < 4; ++gi) {
        FragB16 b = load_fragB_lds(lds_lane, gi * 16 + kt);
        acc[gi] = wmma_bf16(a, b, acc[gi]);
      }
    }
    __bf16* hout = p.hbuf + (size_t)(cur ^ 1) * (BQ * HID);
#pragma unroll
    for (int r = 0; r < 8; ++r) {
      int b = brow0 + r + (hi << 3);
      float iv = sigf(acc[0][r]);
      float gv = tanhfast(acc[2][r]);
      float ov = sigf(acc[3][r]);
      float cn = iv * gv;                 // c = i*g (f*0 dropped)
      float hv = ov * tanhfast(cn);
      hout[(size_t)b * HID + jcol] = (__bf16)hv;
      p.fin[((size_t)b * TOUT + t) * (2 * HID) + jcol] = hv;  // h half of concat
    }
    grid_barrier(p.bar_cnt, p.bar_gen);

    // attention scores s[b,n] = <cl[b,n,:], h_t[b,:]>  (bf16 from L2)
    const __bf16* ht = p.hbuf + (size_t)(cur ^ 1) * (BQ * HID);
    for (int idx = gtid; idx < BQ * NSEQ; idx += NBLOCKS * NTHREADS) {
      int b = idx >> 10, nn = idx & (NSEQ - 1);
      const uint4* hr = (const uint4*)(ht + (size_t)b * HID);
      const uint4* cr = (const uint4*)(p.cl + ((size_t)b * NSEQ + nn) * HID);
      float s = 0.f;
      for (int k = 0; k < HID / 8; ++k) s = dot8(hr[k], cr[k], s);
      p.scores[idx] = s;
    }
    grid_barrier(p.bar_cnt, p.bar_gen);

    // softmax + context: 2 batches per block (block-local, no grid barrier)
    for (int bi = 0; bi < 2; ++bi) {
      int b = blockIdx.x * 2 + bi;
      float* srow = p.scores + (size_t)b * NSEQ;
      float lm = -1e30f;
      for (int k = tid; k < NSEQ; k += NTHREADS) lm = fmaxf(lm, srow[k]);
      float mx = block_reduce_max(lm, lds_red);
      float ls = 0.f;
      for (int k = tid; k < NSEQ; k += NTHREADS) {
        float e = __expf(srow[k] - mx);
        srow[k] = e;
        ls += e;
      }
      float inv = 1.f / block_reduce_sum(ls, lds_red);

      float a0 = 0.f, a1 = 0.f, a2 = 0.f, a3 = 0.f;
      const __bf16* clb = p.cl + (size_t)b * NSEQ * HID + tid * 4;
      for (int nn = 0; nn < NSEQ; ++nn) {
        float cf = srow[nn] * inv;
        uint2 cu = *(const uint2*)(clb + (size_t)nn * HID);
        float2 x0 = bf2f2(cu.x), x1 = bf2f2(cu.y);
        a0 = fmaf(cf, x0.x, a0); a1 = fmaf(cf, x0.y, a1);
        a2 = fmaf(cf, x1.x, a2); a3 = fmaf(cf, x1.y, a3);
      }
      float* fo = p.fin + ((size_t)b * TOUT + t) * (2 * HID) + HID + tid * 4;
      fo[0] = a0; fo[1] = a1; fo[2] = a2; fo[3] = a3;
      __syncthreads();
    }
    cur ^= 1;
  }

  grid_barrier(p.bar_cnt, p.bar_gen);

  // output projection: out = leaky_relu(fin) @ W_out^T + b_out  ([B,T,C], tiny)
  for (int idx = gtid; idx < BQ * TOUT * CIN; idx += NBLOCKS * NTHREADS) {
    int cc = idx & (CIN - 1);
    int btq = idx >> 5;                     // b*TOUT + t
    const float* fr = p.fin + (size_t)btq * (2 * HID);
    const float* wr = p.W_out + (size_t)cc * (2 * HID);
    float s = p.b_out[cc];
    for (int k = 0; k < 2 * HID; ++k) {
      float x = fr[k];
      x = x > 0.f ? x : 0.01f * x;
      s = fmaf(x, wr[k], s);
    }
    p.out[idx] = s;
  }
}

// ---------------- host launch ----------------

extern "C" void kernel_launch(void* const* d_in, const int* in_sizes, int n_in,
                              void* d_out, int out_size, void* d_ws, size_t ws_size,
                              hipStream_t stream) {
  (void)in_sizes; (void)n_in; (void)out_size; (void)ws_size;
  char* ws = (char*)d_ws;
  size_t off = 0;
  auto take = [&](size_t bytes) -> char* {
    char* r = ws + off;
    off = (off + bytes + 255) & ~(size_t)255;
    return r;
  };

  Ptrs p;
  p.cond     = (const float*)d_in[0];
  p.Wih_enc  = (const float*)d_in[1];
  p.Whh_enc  = (const float*)d_in[2];
  p.bih_enc  = (const float*)d_in[3];
  p.bhh_enc  = (const float*)d_in[4];
  p.Wih_cell = (const float*)d_in[5];
  // d_in[6] (Whh_cell) unused: decoder cell runs with zero (h,c) state
  p.bih_cell = (const float*)d_in[7];
  p.bhh_cell = (const float*)d_in[8];
  p.W_out    = (const float*)d_in[9];
  p.b_out    = (const float*)d_in[10];
  p.out      = (float*)d_out;

  unsigned int* bar = (unsigned int*)take(64);
  p.bar_cnt = bar;
  p.bar_gen = bar + 1;
  p.bias_enc   = (float*)take((size_t)G4H * 4);
  p.bias_cell  = (float*)take((size_t)G4H * 4);
  p.hbuf       = (__bf16*)take((size_t)2 * BQ * HID * 2);   // double buffer, one pointer
  p.cond_bf    = (__bf16*)take((size_t)BQ * NSEQ * CIN * 2);
  p.whh_bf     = (__bf16*)take((size_t)G4H * HID * 2);
  p.wih_enc_bf = (__bf16*)take((size_t)G4H * CIN * 2);
  p.wih_cell_bf= (__bf16*)take((size_t)G4H * HID * 2);
  p.scores     = (float*)take((size_t)BQ * NSEQ * 4);
  p.fin        = (float*)take((size_t)BQ * TOUT * 2 * HID * 4);
  p.cl         = (__bf16*)take((size_t)BQ * NSEQ * HID * 2);   // 128MB, L2-resident

  (void)hipFuncSetAttribute((const void*)ccg_persistent_kernel,
                            hipFuncAttributeMaxDynamicSharedMemorySize,
                            (int)SMEM_BYTES);

  ccg_init_kernel<<<512, 256, 0, stream>>>(p);
  ccg_persistent_kernel<<<NBLOCKS, NTHREADS, SMEM_BYTES, stream>>>(p);
}
